// SPVCNN_86002425135827
// MI455X (gfx1250) — compile-verified, matrix operand order checked by reading, not run on hardware
//
#include <hip/hip_runtime.h>
#include <stdint.h>

typedef __bf16 bf16_t;
typedef __attribute__((ext_vector_type(16))) __bf16 v16bf;
typedef __attribute__((ext_vector_type(8)))  __bf16 v8bf;
typedef __attribute__((ext_vector_type(8)))  float  v8f;
typedef __attribute__((ext_vector_type(4)))  unsigned int u32x4;
typedef __attribute__((ext_vector_type(4)))  int i32x4;
typedef __attribute__((ext_vector_type(8)))  int i32x8;

#if defined(__gfx1250__) && __has_builtin(__builtin_amdgcn_tensor_load_to_lds)
#define SPV_HAS_TDM 1
#else
#define SPV_HAS_TDM 0
#endif

#define DPK 1024LL
#define SENTK 0x7FFFFFFFFFFFFFFFLL

#define MODE_CONV3  0
#define MODE_DOWN   1
#define MODE_DECONV 2
#define MODE_LIN    3

// ---------------- device helpers ----------------

__device__ __forceinline__ long long dev_pack(int x, int y, int z, int b) {
  return (((long long)b * DPK + (long long)x) * DPK + (long long)y) * DPK + (long long)z;
}

__device__ __forceinline__ void dev_unpack(long long k, int& x, int& y, int& z, int& b) {
  long long t = k;
  z = (int)(t % DPK); t /= DPK;
  y = (int)(t % DPK); t /= DPK;
  x = (int)(t % DPK); b = (int)(t / DPK);
}

__device__ __forceinline__ int dev_bsearch(const long long* __restrict__ keys, int n, long long q) {
  int lo = 0, hi = n;
  while (lo < hi) { int mid = (lo + hi) >> 1; if (keys[mid] < q) lo = mid + 1; else hi = mid; }
  if (lo < n && keys[lo] == q) return lo;
  return -1;
}

__device__ __forceinline__ unsigned short f2us_bf16(float f) {
  union { float f; unsigned u; } x; x.f = f;
  return (unsigned short)((x.u + 0x7FFFu + ((x.u >> 16) & 1u)) >> 16);
}

union V16 { v16bf v; v8bf h[2]; };

// A-fragment loader: ISA 16-bit A layout = two contiguous 16B runs of the
// gathered bf16 feature row (lanes 0-15: K half*8..+7 & 16+half*8..+7).
__device__ __forceinline__ V16 spv_load_a(const bf16_t* __restrict__ fin, int Cin,
                                          int idx, bool amask, int kc, int half) {
  V16 a;
#pragma unroll
  for (int e = 0; e < 16; e++) a.v[e] = __builtin_bit_cast(bf16_t, (unsigned short)0);
  if (!amask) {
    const bf16_t* rp = fin + (long long)idx * Cin + kc;
    if (kc + 32 <= Cin) {
      a.h[0] = *(const v8bf*)(rp + half * 8);
      a.h[1] = *(const v8bf*)(rp + 16 + half * 8);
    } else {
#pragma unroll
      for (int e = 0; e < 16; e++) {
        int kloc = (e < 8) ? (half * 8 + e) : (16 + half * 8 + (e - 8));
        if (kc + kloc < Cin) a.v[e] = rp[kloc];
      }
    }
  }
  return a;
}

// ---------------- geometry kernels ----------------

__global__ void spv_pack_pts(const int* __restrict__ coords, const int* __restrict__ cb,
                             int N, int n2, long long* __restrict__ keys) {
  int i = blockIdx.x * 256 + threadIdx.x;
  if (i >= n2) return;
  keys[i] = (i < N) ? dev_pack(coords[3 * i], coords[3 * i + 1], coords[3 * i + 2], cb[i]) : SENTK;
}

__global__ void spv_pack_quant(const long long* __restrict__ uk, const int* __restrict__ np,
                               int s, int n2, long long* __restrict__ keys) {
  int i = blockIdx.x * 256 + threadIdx.x;
  if (i >= n2) return;
  int n = *np;
  if (i < n) {
    int x, y, z, b; dev_unpack(uk[i], x, y, z, b);
    keys[i] = dev_pack((x / s) * s, (y / s) * s, (z / s) * s, b);
  } else {
    keys[i] = SENTK;
  }
}

__global__ void spv_bitonic(long long* __restrict__ keys, int n, int j, int k) {
  int i = blockIdx.x * 256 + threadIdx.x;
  int ixj = i ^ j;
  if (i < n && ixj > i && ixj < n) {
    bool up = ((i & k) == 0);
    long long a = keys[i], b = keys[ixj];
    if ((a > b) == up) { keys[i] = b; keys[ixj] = a; }
  }
}

__global__ void spv_mark(const long long* __restrict__ keys, int n2, int* __restrict__ flags) {
  int i = blockIdx.x * 256 + threadIdx.x;
  if (i >= n2) return;
  long long k = keys[i];
  flags[i] = (k != SENTK && (i == 0 || keys[i - 1] != k)) ? 1 : 0;
}

__global__ void spv_scan_block(const int* __restrict__ flags, int* __restrict__ partial,
                               int* __restrict__ bsum, int n2) {
  __shared__ int sh[256];
  int tid = threadIdx.x;
  int i = blockIdx.x * 256 + tid;
  int v = (i < n2) ? flags[i] : 0;
  sh[tid] = v; __syncthreads();
  for (int ofs = 1; ofs < 256; ofs <<= 1) {
    int t = (tid >= ofs) ? sh[tid - ofs] : 0;
    __syncthreads();
    sh[tid] += t;
    __syncthreads();
  }
  if (i < n2) partial[i] = sh[tid];
  if (tid == 255) bsum[blockIdx.x] = sh[255];
}

__global__ void spv_scan_sums(int* __restrict__ bsum, int nb) {
  __shared__ int sh[1024];
  int tid = threadIdx.x;
  int v = (tid < nb) ? bsum[tid] : 0;
  sh[tid] = v; __syncthreads();
  for (int ofs = 1; ofs < 1024; ofs <<= 1) {
    int t = (tid >= ofs) ? sh[tid - ofs] : 0;
    __syncthreads();
    sh[tid] += t;
    __syncthreads();
  }
  if (tid < nb) bsum[tid] = sh[tid];
}

__global__ void spv_scan_compact(const long long* __restrict__ keys, const int* __restrict__ flags,
                                 const int* __restrict__ partial, const int* __restrict__ bsum,
                                 int n2, long long* __restrict__ ukeys, int* __restrict__ count) {
  int i = blockIdx.x * 256 + threadIdx.x;
  if (i >= n2) return;
  int pre = (blockIdx.x > 0) ? bsum[blockIdx.x - 1] : 0;
  int incl = partial[i] + pre;
  if (flags[i]) ukeys[incl - 1] = keys[i];
  if (i == n2 - 1) *count = incl;
}

// ---------------- small utility kernels ----------------

__global__ void spv_cvt_bf16(const float* __restrict__ src, bf16_t* __restrict__ dst, long long n) {
  long long i = (long long)blockIdx.x * 256 + threadIdx.x;
  if (i < n) dst[i] = __builtin_bit_cast(bf16_t, f2us_bf16(src[i]));
}

// Relayout conv weights (K, Cin, Cout) f32 -> bf16 fragment order:
// dst[((k*Kc32 + kc32)*Cp + c)*32 + kl] with K zero-padded to 32 and Cout padded to Cp=ceil128.
// A wave's B-fragment is then two contiguous 16B runs per lane, and a block's
// (k, kc32) B-tile (128 cols x 32 K) is one contiguous 8KB region (TDM-friendly).
__global__ void spv_w_relayout(const float* __restrict__ w, bf16_t* __restrict__ dst,
                               int KOFF, int Cin, int Cout) {
  int Kc32 = (Cin + 31) >> 5;
  int Cp = (Cout + 127) & ~127;
  long long tot = (long long)KOFF * Kc32 * Cp * 32;
  long long i = (long long)blockIdx.x * 256 + threadIdx.x;
  if (i >= tot) return;
  int kl = (int)(i & 31);
  long long t = i >> 5;
  int c = (int)(t % Cp); t /= Cp;
  int kc32 = (int)(t % Kc32);
  int k = (int)(t / Kc32);
  int kr = kc32 * 32 + kl;
  float v = (kr < Cin && c < Cout) ? w[((long long)k * Cin + kr) * Cout + c] : 0.f;
  dst[i] = __builtin_bit_cast(bf16_t, f2us_bf16(v));
}

__global__ void spv_concat_bf16(const float* __restrict__ a, int Ca,
                                const float* __restrict__ b, int Cb,
                                bf16_t* __restrict__ out, long long rows) {
  long long i = (long long)blockIdx.x * 256 + threadIdx.x;
  int C = Ca + Cb;
  long long tot = rows * (long long)C;
  if (i >= tot) return;
  long long r = i / C; int c = (int)(i % C);
  float v = (c < Ca) ? a[r * Ca + c] : b[r * Cb + (c - Ca)];
  out[i] = __builtin_bit_cast(bf16_t, f2us_bf16(v));
}

__global__ void spv_bn_fold(const float* __restrict__ g, const float* __restrict__ b,
                            const float* __restrict__ m, const float* __restrict__ v,
                            const float* __restrict__ linb,
                            float* __restrict__ scale, float* __restrict__ bias, int C) {
  int c = blockIdx.x * 64 + threadIdx.x;
  if (c >= C) return;
  float sc = g[c] * rsqrtf(v[c] + 1e-5f);
  float lb = linb ? linb[c] : 0.f;
  scale[c] = sc;
  bias[c]  = sc * (lb - m[c]) + b[c];
}

__global__ void spv_add_relu(const float* __restrict__ a, const float* __restrict__ b,
                             float* __restrict__ out, const int* __restrict__ np, int C) {
  long long i = (long long)blockIdx.x * 256 + threadIdx.x;
  long long tot = (long long)(*np) * C;
  if (i >= tot) return;
  out[i] = fmaxf(a[i] + b[i], 0.f);
}

// ---------------- voxelize / devoxelize ----------------

__global__ void spv_vox_scatter(const float* __restrict__ fp, int C,
                                const long long* __restrict__ keys, const int* __restrict__ np, int s,
                                const int* __restrict__ zxyz, const int* __restrict__ zb, int P,
                                float* __restrict__ sums, float* __restrict__ cnt) {
  int p = blockIdx.x * 256 + threadIdx.x;
  if (p >= P) return;
  int n = *np;
  int x = (zxyz[3 * p] / s) * s;
  int y = (zxyz[3 * p + 1] / s) * s;
  int z = (zxyz[3 * p + 2] / s) * s;
  int id = dev_bsearch(keys, n, dev_pack(x, y, z, zb[p]));
  if (id < 0) return;
  for (int c = 0; c < C; c++)
    atomicAdd(&sums[(long long)id * C + c], fp[(long long)p * C + c]);
  atomicAdd(&cnt[id], 1.0f);
}

__global__ void spv_vox_final(float* __restrict__ fv, const float* __restrict__ cnt,
                              const int* __restrict__ np, int C) {
  long long i = (long long)blockIdx.x * 256 + threadIdx.x;
  long long tot = (long long)(*np) * C;
  if (i >= tot) return;
  long long r = i / C;
  fv[i] = fv[i] / fmaxf(cnt[r], 1.0f);
}

__global__ void spv_devox(const float* __restrict__ fv, int C,
                          const long long* __restrict__ keys, const int* __restrict__ np, int s,
                          const int* __restrict__ zxyz, const int* __restrict__ zb, int P,
                          float* __restrict__ out, int accum) {
  int p = blockIdx.x * 256 + threadIdx.x;
  if (p >= P) return;
  int n = *np;
  float inv = 1.0f / (float)s;
  float gx = zxyz[3 * p] * inv, gy = zxyz[3 * p + 1] * inv, gz = zxyz[3 * p + 2] * inv;
  float bx = floorf(gx), by = floorf(gy), bz = floorf(gz);
  int ix = (int)bx, iy = (int)by, iz = (int)bz;
  float fx = gx - bx, fy = gy - by, fz = gz - bz;
  int b = zb[p];
  int idx8[8]; float w8[8];
#pragma unroll
  for (int t = 0; t < 8; t++) {
    int oi = (t >> 2) & 1, oj = (t >> 1) & 1, ok = t & 1;
    long long q = dev_pack((ix + oi) * s, (iy + oj) * s, (iz + ok) * s, b);
    int id = dev_bsearch(keys, n, q);
    float w = (oi ? fx : 1.f - fx) * (oj ? fy : 1.f - fy) * (ok ? fz : 1.f - fz);
    if (id < 0) { w = 0.f; id = 0; }
    idx8[t] = id; w8[t] = w;
  }
  for (int c = 0; c < C; c++) {
    float a = 0.f;
#pragma unroll
    for (int t = 0; t < 8; t++) a += w8[t] * fv[(long long)idx8[t] * C + c];
    long long o = (long long)p * C + c;
    out[o] = accum ? (out[o] + a) : a;
  }
}

__global__ void spv_out_gather(const float* __restrict__ z3, int C,
                               const int* __restrict__ batch, const int* __restrict__ v2pid,
                               const int* __restrict__ vnum, int B, int P,
                               float* __restrict__ out) {
  int p = blockIdx.x * 256 + threadIdx.x;
  if (p >= P) return;
  int b = batch[p];
  int inc = 0;
  for (int j = 0; j < b && j < B; j++) inc += vnum[j];
  int iv = v2pid[p] + inc;
  for (int c = 0; c < C; c++) out[(long long)p * C + c] = z3[(long long)iv * C + c];
}

// ---------------- gather-GEMM sparse conv with bf16 WMMA ----------------
//
// Block: 256 threads = 8 wave32s; block tile = 32 output voxels x 128 cols.
// Each wave owns 16 cols and computes TWO 16x16 WMMA tiles (rows row0..+15
// and row0+16..+31) sharing one B fragment -> 2x FLOPs per B fetch and per
// neighbor search. Neighbor indices computed once per block (one wave, LDS).
// A fragments: two global_load_b128 per lane straight from gathered bf16
// feature rows. B: block's 8KB tile staged to LDS by the Tensor Data Mover
// (tensor_load_to_lds + s_wait_tensorcnt) when available, else direct b128
// loads. f32 accumulation via v_wmma_f32_16x16x32_bf16; BN+ReLU fused.

__global__ void __launch_bounds__(256)
spv_spconv_wmma(const bf16_t* __restrict__ fin,
                const long long* __restrict__ keys_in, const int* __restrict__ n_in_p, int n_in_c,
                const long long* __restrict__ keys_out, const int* __restrict__ n_out_p, int n_out_c,
                const bf16_t* __restrict__ Wt, int Cin, int Cout,
                const float* __restrict__ bn_scale, const float* __restrict__ bn_bias,
                float* __restrict__ fout, int mode, int sstr, int relu) {
  int n_out = n_out_p ? *n_out_p : n_out_c;
  int n_in  = n_in_p  ? *n_in_p  : n_in_c;
  int row0 = blockIdx.x * 32;
  if (row0 >= n_out) return;   // uniform per block

  int wave = threadIdx.x >> 5;
  int lane = threadIdx.x & 31;
  int colB = blockIdx.y * 128;
  int col0 = colB + wave * 16;

  int Kc32 = (Cin + 31) >> 5;
  int Cp = (Cout + 127) & ~127;

  __shared__ int s_idx[32];
  __shared__ int s_didx[32];
#if SPV_HAS_TDM
  __shared__ bf16_t Bs[4096];   // 128 cols x 32 K bf16 = 8KB, filled by TDM
#endif

  v8f acc0 = {};
  v8f acc1 = {};
  int KOFF = (mode == MODE_CONV3) ? 27 : ((mode == MODE_LIN) ? 1 : 8);

  int m = lane & 15;
  int half = lane >> 4;
  int c = col0 + m;
  if (c >= Cout) c = Cout - 1;   // padded weight cols are zero anyway

  for (int k = 0; k < KOFF; k++) {
    __syncthreads();
    if (threadIdx.x < 32) {
      int r = row0 + (int)threadIdx.x;
      int idx = -1, dd = 0;
      if (r < n_out) {
        if (mode == MODE_LIN) {
          idx = r;
        } else {
          int x, y, z, b; dev_unpack(keys_out[r], x, y, z, b);
          if (mode == MODE_CONV3) {
            int ox = k / 9 - 1, oy = (k / 3) % 3 - 1, oz = k % 3 - 1;
            idx = dev_bsearch(keys_in, n_in, dev_pack(x + sstr * ox, y + sstr * oy, z + sstr * oz, b));
          } else if (mode == MODE_DOWN) {
            int ox = (k >> 2) & 1, oy = (k >> 1) & 1, oz = k & 1;
            idx = dev_bsearch(keys_in, n_in, dev_pack(x + sstr * ox, y + sstr * oy, z + sstr * oz, b));
          } else { // MODE_DECONV, sstr = coarse stride S
            int S = sstr, sf = S >> 1;
            int qx = (x / S) * S, qy = (y / S) * S, qz = (z / S) * S;
            idx = dev_bsearch(keys_in, n_in, dev_pack(qx, qy, qz, b));
            dd = (((x - qx) / sf) * 2 + (y - qy) / sf) * 2 + (z - qz) / sf;
          }
        }
      }
      s_idx[threadIdx.x] = idx;
      s_didx[threadIdx.x] = dd;
    }
    __syncthreads();

    int idx0 = s_idx[m];
    int idx1 = s_idx[16 + m];
    bool msk0 = (idx0 < 0) || (mode == MODE_DECONV && s_didx[m] != k);
    bool msk1 = (idx1 < 0) || (mode == MODE_DECONV && s_didx[16 + m] != k);

    for (int kc32 = 0; kc32 < Kc32; kc32++) {
      int kc = kc32 << 5;

#if SPV_HAS_TDM
      __syncthreads();                    // previous B-tile fully consumed
      if (wave == 0) {
        const bf16_t* gsrc = Wt + (((long long)(k * Kc32 + kc32) * Cp + colB) << 5);
        unsigned long long ga = (unsigned long long)(uintptr_t)gsrc;
        u32x4 g0;
        g0[0] = 1u;                                        // count=1 valid descriptor
        g0[1] = (unsigned)(uintptr_t)(&Bs[0]);             // LDS destination
        g0[2] = (unsigned)(ga & 0xFFFFFFFFull);            // global_addr[31:0]
        g0[3] = (unsigned)((ga >> 32) & 0x1FFFFFFull) | (2u << 30); // addr[56:32] | type=2
        i32x8 g1;
        g1[0] = 0x00010000;       // data_size = 1 (2 bytes)
        g1[1] = (int)0x10000000;  // tensor_dim0 = 4096 (bits 63:48)
        g1[2] = 0x00010000;       // tensor_dim0 hi = 0, tensor_dim1 = 1
        g1[3] = (int)0x10000000;  // tile_dim0 = 4096 (bits 127:112)
        g1[4] = 0;                // tile_dim1/2 unused
        g1[5] = 4096;             // tensor_dim0_stride
        g1[6] = 0;
        g1[7] = 0;
        i32x4 gz = {0, 0, 0, 0};
#if __clang_major__ >= 23
        i32x8 gz8 = {0, 0, 0, 0, 0, 0, 0, 0};
        __builtin_amdgcn_tensor_load_to_lds(g0, g1, gz, gz, gz8, 0);
#else
        __builtin_amdgcn_tensor_load_to_lds(g0, g1, gz, gz, 0);
#endif
#if __has_builtin(__builtin_amdgcn_s_wait_tensorcnt)
        __builtin_amdgcn_s_wait_tensorcnt(0);
#endif
      }
      __syncthreads();                    // B-tile visible to all waves
#endif

      // ---- A fragments for the two row tiles ----
      V16 a0 = spv_load_a(fin, Cin, idx0, msk0, kc, half);
      V16 a1 = spv_load_a(fin, Cin, idx1, msk1, kc, half);

      // ---- B fragment: two contiguous 16B runs (LDS via TDM, or global) ----
      V16 b;
      {
#if SPV_HAS_TDM
        const bf16_t* bp = &Bs[(c - colB) * 32];
#else
        const bf16_t* bp = Wt + (((long long)(k * Kc32 + kc32) * Cp + c) << 5);
        if (kc32 + 1 < Kc32)
          __builtin_prefetch(Wt + (((long long)(k * Kc32 + kc32 + 1) * Cp + c) << 5), 0, 0);
#endif
        b.h[0] = *(const v8bf*)(bp + half * 8);
        b.h[1] = *(const v8bf*)(bp + 16 + half * 8);
      }

#if defined(__gfx1250__)
      acc0 = __builtin_amdgcn_wmma_f32_16x16x32_bf16(false, a0.v, false, b.v,
                                                     (short)0, acc0, false, false);
      acc1 = __builtin_amdgcn_wmma_f32_16x16x32_bf16(false, a1.v, false, b.v,
                                                     (short)0, acc1, false, false);
#endif
    }
  }

  // epilogue: C layout VGPR r -> M=r (lanes 0-15) / M=r+8 (lanes 16-31), N=lane&15
  int cc = col0 + m;
  int mo = half * 8;
  if (cc < Cout) {
    float s = bn_scale[cc], bsh = bn_bias[cc];
#pragma unroll
    for (int r = 0; r < 8; r++) {
      int m0 = row0 + mo + r;
      if (m0 < n_out) {
        float v = acc0[r] * s + bsh;
        if (relu) v = fmaxf(v, 0.f);
        fout[(long long)m0 * Cout + cc] = v;
      }
      int m1 = row0 + 16 + mo + r;
      if (m1 < n_out) {
        float v = acc1[r] * s + bsh;
        if (relu) v = fmaxf(v, 0.f);
        fout[(long long)m1 * Cout + cc] = v;
      }
    }
  }
}

// ---------------- host orchestration ----------------

extern "C" void kernel_launch(void* const* d_in, const int* in_sizes, int n_in,
                              void* d_out, int out_size, void* d_ws, size_t ws_size,
                              hipStream_t stream) {
  const int*   coords = (const int*)d_in[0];
  const int*   cbatch = (const int*)d_in[1];
  const float* vx     = (const float*)d_in[2];
  const int*   vnum   = (const int*)d_in[3];
  const int*   batch  = (const int*)d_in[4];
  const int*   v2pid  = (const int*)d_in[5];

  int N = in_sizes[0] / 3;
  int B = in_sizes[3];
  int P = in_sizes[4];
  int n2 = 1; while (n2 < N) n2 <<= 1;
  int M = (N > P) ? N : P;

  // ---- param walk (jax pytree flatten order: dict keys sorted) ----
  int pi = 6;
  auto nf = [&]() -> const float* { return (const float*)d_in[pi++]; };
  struct BN  { const float *b, *g, *m, *v; };
  struct Res { BN bn1, bn2, bnd; const float *w1, *w2, *wd; int has; };
  struct Lin { const float* b; BN bn; const float* w; };
  struct Stage { BN bn; Res r1, r2; const float* w; };
  auto getBN = [&]() { BN x; x.b = nf(); x.g = nf(); x.m = nf(); x.v = nf(); return x; };
  auto getLin = [&]() { Lin l; l.b = nf(); l.bn = getBN(); l.w = nf(); return l; };
  auto getRes = [&](bool wd) {
    Res r; r.bn1 = getBN(); r.bn2 = getBN();
    if (wd) r.bnd = getBN();
    r.w1 = nf(); r.w2 = nf();
    if (wd) r.wd = nf();
    r.has = wd ? 1 : 0;
    return r;
  };
  auto getStage = [&](bool wd1) {
    Stage s; s.bn = getBN(); s.r1 = getRes(wd1); s.r2 = getRes(false); s.w = nf(); return s;
  };
  Lin pt0 = getLin(), pt1 = getLin(), pt2 = getLin();
  Stage s1 = getStage(false), s2 = getStage(true), s3 = getStage(true), s4 = getStage(true);
  BN stem_bn1 = getBN(), stem_bn2 = getBN();
  const float* stem_w1 = nf();
  const float* stem_w2 = nf();
  Stage u1 = getStage(true), u2 = getStage(true), u3 = getStage(true), u4 = getStage(true);

  // ---- workspace carve ----
  size_t off = 0;
  char* base = (char*)d_ws;
  auto alloc = [&](size_t bytes) -> void* {
    void* p = base + off;
    off = (off + bytes + 255) & ~(size_t)255;
    return p;
  };
  long long* keys_pad = (long long*)alloc((size_t)n2 * 8);
  int* flags   = (int*)alloc((size_t)n2 * 4);
  int* partial = (int*)alloc((size_t)n2 * 4);
  int* bsum    = (int*)alloc(4096 * 4);
  long long* uk[5];
  for (int l = 0; l < 5; l++) uk[l] = (long long*)alloc((size_t)N * 8);
  int* nv = (int*)alloc(64);
  float* va = (float*)alloc((size_t)M * 256 * 4);
  float* vb = (float*)alloc((size_t)M * 256 * 4);
  float* vc = (float*)alloc((size_t)M * 256 * 4);
  float* vd = (float*)alloc((size_t)M * 256 * 4);
  bf16_t* ba = (bf16_t*)alloc((size_t)M * 384 * 2);
  bf16_t* bb = (bf16_t*)alloc((size_t)M * 384 * 2);
  float* cnt = (float*)alloc((size_t)M * 4);
  float* xs0 = (float*)alloc((size_t)M * 32 * 4);
  float* xs1 = (float*)alloc((size_t)M * 32 * 4);
  float* xs2 = (float*)alloc((size_t)M * 64 * 4);
  float* xs3 = (float*)alloc((size_t)M * 128 * 4);
  float* xs4 = (float*)alloc((size_t)M * 256 * 4);
  float* pA  = (float*)alloc((size_t)M * 256 * 4);
  float* pB  = (float*)alloc((size_t)M * 256 * 4);
  float* pC  = (float*)alloc((size_t)M * 256 * 4);

  int nb = (n2 + 255) / 256;

  // ---- helpers ----
  auto sortpad = [&]() {
    for (int k = 2; k <= n2; k <<= 1)
      for (int j = k >> 1; j > 0; j >>= 1)
        spv_bitonic<<<nb, 256, 0, stream>>>(keys_pad, n2, j, k);
  };
  auto uniq = [&](long long* dst, int* cptr) {
    spv_mark<<<nb, 256, 0, stream>>>(keys_pad, n2, flags);
    spv_scan_block<<<nb, 256, 0, stream>>>(flags, partial, bsum, n2);
    spv_scan_sums<<<1, 1024, 0, stream>>>(bsum, nb);
    spv_scan_compact<<<nb, 256, 0, stream>>>(keys_pad, flags, partial, bsum, n2, dst, cptr);
  };
  auto wrl = [&](const float* w, int koff, int Cin, int Cout) -> bf16_t* {
    int Kc32 = (Cin + 31) >> 5;
    int Cp = (Cout + 127) & ~127;
    long long n = (long long)koff * Kc32 * Cp * 32;
    bf16_t* d = (bf16_t*)alloc((size_t)n * 2);
    spv_w_relayout<<<dim3((unsigned)((n + 255) / 256)), 256, 0, stream>>>(w, d, koff, Cin, Cout);
    return d;
  };
  auto fold = [&](const BN& bn, const float* lb, int C) -> float* {
    float* s = (float*)alloc((size_t)2 * C * 4);
    spv_bn_fold<<<(C + 63) / 64, 64, 0, stream>>>(bn.g, bn.b, bn.m, bn.v, lb, s, s + C, C);
    return s;
  };
  auto cvtf = [&](const float* src, bf16_t* dst, long long n) {
    spv_cvt_bf16<<<dim3((unsigned)((n + 255) / 256)), 256, 0, stream>>>(src, dst, n);
  };
  auto gemm = [&](const bf16_t* fin,
                  const long long* kin, const int* nip, int nic,
                  const long long* kout, const int* nop, int noc, int ncap,
                  const bf16_t* W, int Cin, int Cout, const float* sb,
                  float* fo, int mode, int sstr, int relu) {
    dim3 g((ncap + 31) / 32, (Cout + 127) / 128);
    spv_spconv_wmma<<<g, 256, 0, stream>>>(fin, kin, nip, nic, kout, nop, noc,
                                           W, Cin, Cout, sb, sb + Cout, fo, mode, sstr, relu);
  };
  auto p2v = [&](const float* fp, int C, int l, float* fv) {
    hipMemsetAsync(fv, 0, (size_t)M * C * 4, stream);
    hipMemsetAsync(cnt, 0, (size_t)M * 4, stream);
    spv_vox_scatter<<<(N + 255) / 256, 256, 0, stream>>>(fp, C, uk[l], nv + l, 1 << l,
                                                         coords, cbatch, N, fv, cnt);
    spv_vox_final<<<dim3((unsigned)(((long long)M * C + 255) / 256)), 256, 0, stream>>>(fv, cnt, nv + l, C);
  };
  auto v2p = [&](const float* fv, int C, int l, float* out, int accum) {
    spv_devox<<<(N + 255) / 256, 256, 0, stream>>>(fv, C, uk[l], nv + l, 1 << l,
                                                   coords, cbatch, N, out, accum);
  };
  auto addrelu = [&](const float* a, const float* b, float* o, const int* np, int C) {
    spv_add_relu<<<dim3((unsigned)(((long long)M * C + 255) / 256)), 256, 0, stream>>>(a, b, o, np, C);
  };
  auto resblock = [&](const float* fin, float* t1, float* t2, float* fout,
                      int l, const Res& r, int Cin, int Cout) {
    const long long* K = uk[l]; const int* np = nv + l; int s = 1 << l;
    cvtf(fin, ba, (long long)M * Cin);
    gemm(ba, K, np, 0, K, np, 0, N, wrl(r.w1, 27, Cin, Cout), Cin, Cout,
         fold(r.bn1, nullptr, Cout), t1, MODE_CONV3, s, 1);
    cvtf(t1, bb, (long long)M * Cout);
    gemm(bb, K, np, 0, K, np, 0, N, wrl(r.w2, 27, Cout, Cout), Cout, Cout,
         fold(r.bn2, nullptr, Cout), t2, MODE_CONV3, s, 0);
    const float* scv = fin;
    if (r.has) {
      gemm(ba, K, np, 0, K, np, 0, N, wrl(r.wd, 1, Cin, Cout), Cin, Cout,
           fold(r.bnd, nullptr, Cout), t1, MODE_LIN, 1, 0);
      scv = t1;
    }
    addrelu(t2, scv, fout, np, Cout);
  };
  auto upstage = [&](const float* fin, int li, const Stage& st,
                     int Cin, int Cd, const float* skip, int Cskip, int Cout,
                     float* t1, float* t2, float* t3, float* fout) {
    int lo = li - 1, S = 1 << li, s = 1 << lo;
    const long long* Ki = uk[li]; const int* ni = nv + li;
    const long long* Ko = uk[lo]; const int* no = nv + lo;
    cvtf(fin, ba, (long long)M * Cin);
    gemm(ba, Ki, ni, 0, Ko, no, 0, N, wrl(st.w, 8, Cin, Cd), Cin, Cd,
         fold(st.bn, nullptr, Cd), t1, MODE_DECONV, S, 1);
    int Cc = Cd + Cskip;
    spv_concat_bf16<<<dim3((unsigned)(((long long)M * Cc + 255) / 256)), 256, 0, stream>>>(
        t1, Cd, skip, Cskip, ba, (long long)M);
    // r1 (has wd)
    gemm(ba, Ko, no, 0, Ko, no, 0, N, wrl(st.r1.w1, 27, Cc, Cout), Cc, Cout,
         fold(st.r1.bn1, nullptr, Cout), t2, MODE_CONV3, s, 1);
    cvtf(t2, bb, (long long)M * Cout);
    gemm(bb, Ko, no, 0, Ko, no, 0, N, wrl(st.r1.w2, 27, Cout, Cout), Cout, Cout,
         fold(st.r1.bn2, nullptr, Cout), t3, MODE_CONV3, s, 0);
    gemm(ba, Ko, no, 0, Ko, no, 0, N, wrl(st.r1.wd, 1, Cc, Cout), Cc, Cout,
         fold(st.r1.bnd, nullptr, Cout), t2, MODE_LIN, 1, 0);
    addrelu(t3, t2, t1, no, Cout);
    // r2 (no wd)
    resblock(t1, t2, t3, fout, lo, st.r2, Cout, Cout);
  };

  // ---- build hash levels 0..4 ----
  spv_pack_pts<<<nb, 256, 0, stream>>>(coords, cbatch, N, n2, keys_pad);
  sortpad(); uniq(uk[0], nv + 0);
  for (int l = 1; l < 5; l++) {
    spv_pack_quant<<<nb, 256, 0, stream>>>(uk[l - 1], nv + l - 1, 1 << l, n2, keys_pad);
    sortpad(); uniq(uk[l], nv + l);
  }

  // ---- forward ----
  // initial voxelize: va = p2v(voxel_x C=4, level0)
  p2v(vx, 4, 0, va);
  // stem
  cvtf(va, ba, (long long)M * 4);
  gemm(ba, uk[0], nv + 0, 0, uk[0], nv + 0, 0, N, wrl(stem_w1, 27, 4, 32), 4, 32,
       fold(stem_bn1, nullptr, 32), vb, MODE_CONV3, 1, 1);
  cvtf(vb, ba, (long long)M * 32);
  gemm(ba, uk[0], nv + 0, 0, uk[0], nv + 0, 0, N, wrl(stem_w2, 27, 32, 32), 32, 32,
       fold(stem_bn2, nullptr, 32), xs0, MODE_CONV3, 1, 1);
  // z0 = v2p(x0); f = p2v(z0)
  v2p(xs0, 32, 0, pA, 0);
  p2v(pA, 32, 0, va);

  // down path
  Stage* dstg[4] = { &s1, &s2, &s3, &s4 };
  float* xsb[5]  = { xs0, xs1, xs2, xs3, xs4 };
  int dci[4] = { 32, 32, 64, 128 };
  int dco[4] = { 32, 64, 128, 256 };
  float* f = va;
  for (int i = 1; i <= 4; i++) {
    Stage& st = *dstg[i - 1];
    int ci = dci[i - 1], cd = ci, co = dco[i - 1];
    cvtf(f, ba, (long long)M * ci);
    gemm(ba, uk[i - 1], nv + i - 1, 0, uk[i], nv + i, 0, N, wrl(st.w, 8, ci, cd), ci, cd,
         fold(st.bn, nullptr, cd), vb, MODE_DOWN, 1 << (i - 1), 1);
    resblock(vb, vc, vd, va, i, st.r1, cd, co);
    resblock(va, vc, vd, xsb[i], i, st.r2, co, co);
    f = xsb[i];
  }

  // z1 = ptrans(z0, pt0 32->256) + v2p(x4, level4)
  cvtf(pA, ba, (long long)M * 32);
  gemm(ba, nullptr, nullptr, N, nullptr, nullptr, N, N, wrl(pt0.w, 1, 32, 256), 32, 256,
       fold(pt0.bn, pt0.b, 256), pB, MODE_LIN, 1, 1);
  v2p(xs4, 256, 4, pB, 1);
  // y = p2v(z1, level4)
  p2v(pB, 256, 4, va);

  // u1: level4 -> 3 ; u2: level3 -> 2
  upstage(va, 4, u1, 256, 256, xs3, 128, 256, vb, vc, vd, va);
  upstage(va, 3, u2, 256, 128, xs2, 64, 128, vb, vc, vd, va);

  // z2 = ptrans(z1, pt1 256->128) + v2p(y, level2)
  cvtf(pB, ba, (long long)M * 256);
  gemm(ba, nullptr, nullptr, N, nullptr, nullptr, N, N, wrl(pt1.w, 1, 256, 128), 256, 128,
       fold(pt1.bn, pt1.b, 128), pC, MODE_LIN, 1, 1);
  v2p(va, 128, 2, pC, 1);
  // y = p2v(z2, level2)
  p2v(pC, 128, 2, va);

  // u3: level2 -> 1 ; u4: level1 -> 0
  upstage(va, 2, u3, 128, 96, xs1, 32, 96, vb, vc, vd, va);
  upstage(va, 1, u4, 96, 96, xs0, 32, 96, vb, vc, vd, va);

  // z3 = ptrans(z2, pt2 128->96) + v2p(y, level0)
  cvtf(pC, ba, (long long)M * 128);
  gemm(ba, nullptr, nullptr, N, nullptr, nullptr, N, N, wrl(pt2.w, 1, 128, 96), 128, 96,
       fold(pt2.bn, pt2.b, 96), pA, MODE_LIN, 1, 1);
  v2p(va, 96, 0, pA, 1);

  // final inverse-map gather
  spv_out_gather<<<(P + 255) / 256, 256, 0, stream>>>(pA, 96, batch, v2pid, vnum, B, P, (float*)d_out);

  (void)n_in; (void)out_size; (void)ws_size;
}
